// TransformerBlock_22660247454486
// MI455X (gfx1250) — matrix operand order repaired and checked by reading, not compiled, not run on hardware
//
#include <hip/hip_runtime.h>
#include <hip/hip_bf16.h>

// ---------------------------------------------------------------------------
// CDNA5 (gfx1250) implementation of the EquiformerV2-style block.
// Heavy GEMMs (radial MLP, so2_conv c1/c2, so3_linear, gate/FFN) use
// v_wmma_f32_16x16x32_f16 with M=16 edge/node tiles. Weights are packed into
// per-lane WMMA B-fragment layout once per launch. f16 in / f32 accumulate.
// Sigmoids use v_rcp_f32 (hardware reciprocal) instead of precise division.
// ---------------------------------------------------------------------------

typedef __attribute__((ext_vector_type(16))) _Float16 v16h;
typedef __attribute__((ext_vector_type(8)))  float    v8f;

__device__ const int kLof[9] = {0, 1, 1, 1, 2, 2, 2, 2, 2};

// fast sigmoid: v_exp_f32 + v_add + v_rcp_f32 (no div-fixup chain)
__device__ __forceinline__ float sigf(float x) {
  return __builtin_amdgcn_rcpf(1.f + __expf(-x));
}

__device__ __forceinline__ void atomicMaxF(float* addr, float val) {
  int* ia = (int*)addr;
  int cur = __float_as_int(*addr);
  while (__int_as_float(cur) < val) {
    int old = atomicCAS(ia, cur, __float_as_int(val));
    if (old == cur) break;
    cur = old;
  }
}

// A fragment (16x32 f16): lane = hi*16+m, element e -> K = hi*16+e
__device__ __forceinline__ v16h load_a16(const _Float16* As, int lda) {
  const int lane = threadIdx.x & 31;
  const int m = lane & 15, hi = lane >> 4;
  v16h a;
#pragma unroll
  for (int e = 0; e < 16; ++e) {
    int k = hi * 16 + e;
    a[e] = As[m * lda + k];
  }
  return a;
}

// Block-level GEMM: D[16][NNT*16] (+stacks) = A[16][NKT*32] @ Bpacked, 8 waves.
// B is pre-packed: frag (kt,nt) = 32 lanes x v16h, b[e] = W[kt*32 + hi*16 + e][nt*16 + (lane&15)]
template <int NKT, int NNT, int NSTACKS, typename F>
__device__ __forceinline__ void gemm16(const _Float16* As, int lda, int aStride,
                                       const _Float16* Bpack, const int* bSlice,
                                       int bSliceStrideV, F&& emit) {
  const int lane = threadIdx.x & 31;
  const int wv = threadIdx.x >> 5;
  const v16h* Bp = (const v16h*)Bpack;
  for (int task = wv; task < NSTACKS * NNT; task += 8) {
    int s = task / NNT, nt = task % NNT;
    const v16h* Bs = Bp + (bSlice ? bSlice[s] * bSliceStrideV : 0);
    const _Float16* A = As + s * aStride;
    v8f acc = {};
#pragma unroll
    for (int kt = 0; kt < NKT; ++kt) {
      v16h a = load_a16(A + kt * 32, lda);
      v16h b = Bs[(kt * NNT + nt) * 32 + lane];
      acc = __builtin_amdgcn_wmma_f32_16x16x32_f16(false, a, false, b, (short)0, acc,
                                                   false, false);
    }
    int n = nt * 16 + (lane & 15);
    int hi = lane >> 4;
#pragma unroll
    for (int r = 0; r < 8; ++r) emit(s, hi * 8 + r, n, acc[r]);
  }
}

// Pack f32 weight [Ktot][Ntot] (row-major, in x out) into WMMA B fragments (f16).
__global__ void pack_w_kernel(const float* __restrict__ W, _Float16* __restrict__ out,
                              int Ktot, int Ntot) {
  int total = Ktot * Ntot;
  int nNT = Ntot >> 4;
  for (int p = blockIdx.x * blockDim.x + threadIdx.x; p < total;
       p += gridDim.x * blockDim.x) {
    int e = p & 15;
    int lane = (p >> 4) & 31;
    int frag = p >> 9;
    int kt = frag / nNT, nt = frag - kt * nNT;
    int k = kt * 32 + ((lane >> 4) << 4) + e;
    int n = nt * 16 + (lane & 15);
    out[p] = (_Float16)W[(size_t)k * Ntot + n];
  }
}

__global__ void init_kernel(float* gNode, float* gMx, float* gDen, int nNode) {
  int i = blockIdx.x * blockDim.x + threadIdx.x;
  if (i < nNode * 576) gNode[i] = 0.f;
  if (i < nNode * 4) { gMx[i] = -3.0e38f; gDen[i] = 0.f; }
}

__device__ __forceinline__ float blockSum64(float v, float* red, int c) {
  __syncthreads();
  red[c] = v;
  __syncthreads();
#pragma unroll
  for (int s = 32; s > 0; s >>= 1) {
    if (c < s) red[c] += red[c + s];
    __syncthreads();
  }
  return red[0];
}

// eq_norm: layernorm on l=0 row, RMS-norm on l=1 (rows 1..3) and l=2 (rows 4..8)
__global__ __launch_bounds__(64) void eqnorm_kernel(const float* __restrict__ xin,
                                                    float* __restrict__ out,
                                                    const float* __restrict__ w0,
                                                    const float* __restrict__ b0,
                                                    const float* __restrict__ wl) {
  __shared__ float red[64];
  const int n = blockIdx.x, c = threadIdx.x;
  const float* xp = xin + (size_t)n * 576;
  float* op = out + (size_t)n * 576;
  float v0 = xp[c];
  float mu = blockSum64(v0, red, c) * (1.f / 64.f);
  float d = v0 - mu;
  float var = blockSum64(d * d, red, c) * (1.f / 64.f);
  op[c] = d * rsqrtf(var + 1e-5f) * w0[c] + b0[c];
  float x1 = xp[64 + c], x2 = xp[128 + c], x3 = xp[192 + c];
  float s1 = blockSum64(x1 * x1 + x2 * x2 + x3 * x3, red, c) * (1.f / 192.f);
  float r1 = rsqrtf(s1 + 1e-5f) * wl[c];
  op[64 + c] = x1 * r1; op[128 + c] = x2 * r1; op[192 + c] = x3 * r1;
  float y0 = xp[256 + c], y1 = xp[320 + c], y2 = xp[384 + c], y3 = xp[448 + c],
        y4 = xp[512 + c];
  float s2 = blockSum64(y0 * y0 + y1 * y1 + y2 * y2 + y3 * y3 + y4 * y4, red, c) *
             (1.f / 320.f);
  float r2 = rsqrtf(s2 + 1e-5f) * wl[64 + c];
  op[256 + c] = y0 * r2; op[320 + c] = y1 * r2; op[384 + c] = y2 * r2;
  op[448 + c] = y3 * r2; op[512 + c] = y4 * r2;
}

// Radial MLP: r = (silu(silu(d@w1+b1)@w2+b2))@w3 + b3, per 16-edge tile via WMMA.
__global__ __launch_bounds__(256) void radial_kernel(
    const float* __restrict__ gD, const _Float16* __restrict__ wp1,
    const _Float16* __restrict__ wp2, const _Float16* __restrict__ wp3,
    const float* __restrict__ b1, const float* __restrict__ b2,
    const float* __restrict__ b3, _Float16* __restrict__ gRad) {
  __shared__ _Float16 sX[16 * 64];
  __shared__ _Float16 sT[16 * 64];
  const int t = threadIdx.x, e0 = blockIdx.x * 16;
  for (int i = t; i < 16 * 64; i += 256) sX[i] = (_Float16)gD[(size_t)e0 * 64 + i];
  __syncthreads();
  gemm16<2, 4, 1>(sX, 64, 0, wp1, nullptr, 0, [&](int, int e, int n, float v) {
    float y = v + b1[n]; sT[e * 64 + n] = (_Float16)(y * sigf(y));
  });
  __syncthreads();
  gemm16<2, 4, 1>(sT, 64, 0, wp2, nullptr, 0, [&](int, int e, int n, float v) {
    float y = v + b2[n]; sX[e * 64 + n] = (_Float16)(y * sigf(y));
  });
  __syncthreads();
  gemm16<2, 48, 1>(sX, 64, 0, wp3, nullptr, 0, [&](int, int e, int n, float v) {
    gRad[(size_t)(e0 + e) * 768 + n] = (_Float16)(v + b3[n]);
  });
}

__device__ __forceinline__ void load_hcat(const float* gH1, int sn, int dn, int c,
                                          float hc[9]) {
  const float* hb = (c < 64) ? (gH1 + (size_t)sn * 576 + c)
                             : (gH1 + (size_t)dn * 576 + (c - 64));
#pragma unroll
  for (int j = 0; j < 9; ++j) hc[j] = hb[j * 64];
}

// Pass A: gather+wigner -> so2_conv c1 (WMMA) -> sep_s2_act -> h (f16), attn logits.
__global__ __launch_bounds__(256) void edge_pass_a(
    const float* __restrict__ gH1, const _Float16* __restrict__ gRad,
    const float* __restrict__ gWig, const int* __restrict__ srcp,
    const int* __restrict__ dstp, const _Float16* __restrict__ wpC1W0,
    const _Float16* __restrict__ wpC1M1, const _Float16* __restrict__ wpC1M2,
    const float* __restrict__ c1b0, const float* __restrict__ toG,
    const float* __restrict__ fromG, const float* __restrict__ alnw,
    const float* __restrict__ alnb, const float* __restrict__ adot,
    _Float16* __restrict__ gHmsg, float* __restrict__ gLogits,
    float* __restrict__ gMx) {
  __shared__ float sWig[16 * 81];
  __shared__ _Float16 sA[16 * 512];
  __shared__ _Float16 sHpre[16][576];
  __shared__ _Float16 sYm[2 * 16 * 256];
  __shared__ _Float16 sAF[16][128];
  __shared__ _Float16 sGt[16][64];
  __shared__ float sTG[144], sFG[144];
  __shared__ int sSrc[16], sDst[16];

  const int t = threadIdx.x;
  const int e0 = blockIdx.x * 16;

  for (int i = t; i < 16 * 81; i += 256) sWig[i] = gWig[(size_t)e0 * 81 + i];
  for (int i = t; i < 144; i += 256) { sTG[i] = toG[i]; sFG[i] = fromG[i]; }
  if (t < 16) { sSrc[t] = srcp[e0 + t]; sDst[t] = dstp[e0 + t]; }
  __syncthreads();

  // ---- x0 = (wigner @ [h[src],h[dst]]) rows {0,2,6}, scaled by rad0 ----
  for (int i = t; i < 16 * 128; i += 256) {
    int e = i >> 7, c = i & 127;
    float hc[9];
    load_hcat(gH1, sSrc[e], sDst[e], c, hc);
    const float* wg = sWig + e * 81;
    const _Float16* rd = gRad + (size_t)(e0 + e) * 768;
#pragma unroll
    for (int r = 0; r < 3; ++r) {
      int row = (r == 0) ? 0 : (r == 1 ? 2 : 6);
      float m = 0.f;
#pragma unroll
      for (int j = 0; j < 9; ++j) m += wg[row * 9 + j] * hc[j];
      int col = r * 128 + c;
      sA[e * 384 + col] = (_Float16)(m * (float)rd[col]);
    }
  }
  __syncthreads();

  // ---- y0 = x0 @ c1_w0 + b0 : rows{0,2,6}, alpha_feat, gating ----
  gemm16<12, 24, 1>(sA, 384, 0, wpC1W0, nullptr, 0, [&](int, int e, int n, float v) {
    float y = v + c1b0[n];
    if (n < 192) {
      int rr = n >> 6;
      int row = (rr == 0) ? 0 : ((rr == 1) ? 2 : 6);
      sHpre[e][row * 64 + (n & 63)] = (_Float16)y;
    } else if (n < 320) {
      sAF[e][n - 192] = (_Float16)y;
    } else {
      sGt[e][n - 320] = (_Float16)y;
    }
  });
  __syncthreads();

  // ---- m=1: stacks rows {1,5} / {3,7}, scaled by rad1 ----
  for (int i = t; i < 16 * 128; i += 256) {
    int e = i >> 7, c = i & 127;
    float hc[9];
    load_hcat(gH1, sSrc[e], sDst[e], c, hc);
    const float* wg = sWig + e * 81;
    const _Float16* rd = gRad + (size_t)(e0 + e) * 768 + 384;
#pragma unroll
    for (int r = 0; r < 2; ++r) {
      int rowM = r ? 5 : 1, rowP = r ? 7 : 3;
      float mm = 0.f, mp = 0.f;
#pragma unroll
      for (int j = 0; j < 9; ++j) {
        mm += wg[rowM * 9 + j] * hc[j];
        mp += wg[rowP * 9 + j] * hc[j];
      }
      int col = r * 128 + c;
      float rv = (float)rd[col];
      sA[e * 256 + col] = (_Float16)(mm * rv);
      sA[16 * 256 + e * 256 + col] = (_Float16)(mp * rv);
    }
  }
  __syncthreads();
  gemm16<8, 16, 2>(sA, 256, 16 * 256, wpC1M1, nullptr, 0,
                   [&](int s, int e, int n, float v) {
                     sYm[s * 4096 + e * 256 + n] = (_Float16)v;
                   });
  __syncthreads();
  for (int i = t; i < 16 * 128; i += 256) {
    int e = i >> 7, c = i & 127;
    int r = c >> 6, cc = c & 63;
    float yr0 = (float)sYm[e * 256 + c],        yi0 = (float)sYm[e * 256 + 128 + c];
    float yr1 = (float)sYm[4096 + e * 256 + c], yi1 = (float)sYm[4096 + e * 256 + 128 + c];
    sHpre[e][(r ? 5 : 1) * 64 + cc] = (_Float16)(yr0 - yi1);
    sHpre[e][(r ? 7 : 3) * 64 + cc] = (_Float16)(yr1 + yi0);
  }
  __syncthreads();

  // ---- m=2: stacks rows {4} / {8}, scaled by rad2 ----
  for (int i = t; i < 16 * 128; i += 256) {
    int e = i >> 7, c = i & 127;
    float hc[9];
    load_hcat(gH1, sSrc[e], sDst[e], c, hc);
    const float* wg = sWig + e * 81;
    const _Float16* rd = gRad + (size_t)(e0 + e) * 768 + 640;
    float m4 = 0.f, m8 = 0.f;
#pragma unroll
    for (int j = 0; j < 9; ++j) {
      m4 += wg[4 * 9 + j] * hc[j];
      m8 += wg[8 * 9 + j] * hc[j];
    }
    float rv = (float)rd[c];
    sA[e * 128 + c] = (_Float16)(m4 * rv);
    sA[16 * 128 + e * 128 + c] = (_Float16)(m8 * rv);
  }
  __syncthreads();
  gemm16<4, 8, 2>(sA, 128, 16 * 128, wpC1M2, nullptr, 0,
                  [&](int s, int e, int n, float v) {
                    sYm[s * 4096 + e * 256 + n] = (_Float16)v;
                  });
  __syncthreads();
  for (int i = t; i < 16 * 64; i += 256) {
    int e = i >> 6, cc = i & 63;
    float yr0 = (float)sYm[e * 256 + cc],        yi0 = (float)sYm[e * 256 + 64 + cc];
    float yr1 = (float)sYm[4096 + e * 256 + cc], yi1 = (float)sYm[4096 + e * 256 + 64 + cc];
    sHpre[e][4 * 64 + cc] = (_Float16)(yr0 - yi1);
    sHpre[e][8 * 64 + cc] = (_Float16)(yr1 + yi0);
  }
  __syncthreads();

  // ---- sep_s2_act -> h (store f16) ----
  for (int i = t; i < 16 * 64; i += 256) {
    int e = i >> 6, c = i & 63;
    float gv[16];
#pragma unroll
    for (int g = 0; g < 16; ++g) {
      float s = 0.f;
#pragma unroll
      for (int k = 0; k < 9; ++k) s += sTG[g * 9 + k] * (float)sHpre[e][k * 64 + c];
      gv[g] = s * sigf(s);
    }
    _Float16* ho = gHmsg + (size_t)(e0 + e) * 576;
    float g0 = (float)sGt[e][c];
    ho[c] = (_Float16)(g0 * sigf(g0));
#pragma unroll
    for (int k = 1; k < 9; ++k) {
      float s = 0.f;
#pragma unroll
      for (int g = 0; g < 16; ++g) s += sFG[g * 9 + k] * gv[g];
      ho[k * 64 + c] = (_Float16)s;
    }
  }

  // ---- attention logits: LN -> smooth_leaky_relu -> dot(alpha_dot) ----
  if (t < 64) {
    int e = t >> 2, h = t & 3;
    float mu = 0.f;
#pragma unroll
    for (int c = 0; c < 32; ++c) mu += (float)sAF[e][h * 32 + c];
    mu *= (1.f / 32.f);
    float var = 0.f;
#pragma unroll
    for (int c = 0; c < 32; ++c) {
      float d = (float)sAF[e][h * 32 + c] - mu;
      var += d * d;
    }
    float inv = rsqrtf(var * (1.f / 32.f) + 1e-5f);
    float lg = 0.f;
#pragma unroll
    for (int c = 0; c < 32; ++c) {
      float xn = ((float)sAF[e][h * 32 + c] - mu) * inv * alnw[c] + alnb[c];
      float sl = 0.6f * xn + 0.4f * xn * (2.f * sigf(xn) - 1.f);
      lg += sl * adot[h * 32 + c];
    }
    gLogits[(size_t)(e0 + e) * 4 + h] = lg;
    atomicMaxF(&gMx[(size_t)sDst[e] * 4 + h], lg);
  }
}

__global__ void softmax_edge_kernel(const float* __restrict__ gLogits,
                                    const float* __restrict__ gMx,
                                    const int* __restrict__ dstp,
                                    float* __restrict__ gEx, float* __restrict__ gDen,
                                    int nE) {
  int i = blockIdx.x * blockDim.x + threadIdx.x;
  if (i >= nE * 4) return;
  int e = i >> 2, h = i & 3;
  int d = dstp[e];
  float ex = __expf(gLogits[i] - gMx[d * 4 + h]);
  gEx[i] = ex;
  atomicAdd(&gDen[d * 4 + h], ex);
}

// Pass B: so2_conv c2 (WMMA) -> alpha-scale -> wigner^T -> scatter-add to nodes.
__global__ __launch_bounds__(256) void edge_pass_b(
    const _Float16* __restrict__ gHmsg, const float* __restrict__ gWigG,
    const int* __restrict__ dstp, const _Float16* __restrict__ wpC2W0,
    const _Float16* __restrict__ wpC2M1, const _Float16* __restrict__ wpC2M2,
    const float* __restrict__ c2b0, const float* __restrict__ gEx,
    const float* __restrict__ gDen, float* __restrict__ gNode) {
  __shared__ _Float16 sH[16][576];
  __shared__ _Float16 sA[16 * 256];
  __shared__ _Float16 sV[16][576];
  __shared__ alignas(32) char sUni[16384];  // sYm during GEMMs, wigner afterwards
  __shared__ float sAl[16][4];
  __shared__ int sDst[16];
  _Float16* sYm = (_Float16*)sUni;
  float* sWig = (float*)sUni;

  const int t = threadIdx.x, e0 = blockIdx.x * 16;
  for (int i = t; i < 16 * 576; i += 256) sH[i / 576][i % 576] = gHmsg[(size_t)e0 * 576 + i];
  if (t < 16) sDst[t] = dstp[e0 + t];
  if (t < 64) {
    int e = t >> 2, h = t & 3;
    int d = dstp[e0 + e];
    sAl[e][h] = gEx[(size_t)(e0 + e) * 4 + h] / gDen[(size_t)d * 4 + h];
  }
  __syncthreads();

  // x0 path
  for (int i = t; i < 16 * 192; i += 256) {
    int e = i / 192, n = i % 192;
    int rr = n >> 6, row = (rr == 0) ? 0 : ((rr == 1) ? 2 : 6);
    sA[e * 192 + n] = sH[e][row * 64 + (n & 63)];
  }
  __syncthreads();
  gemm16<6, 12, 1>(sA, 192, 0, wpC2W0, nullptr, 0, [&](int, int e, int n, float v) {
    int rr = n >> 6, row = (rr == 0) ? 0 : ((rr == 1) ? 2 : 6);
    sV[e][row * 64 + (n & 63)] = (_Float16)(v + c2b0[n]);
  });
  __syncthreads();

  // m=1
  for (int i = t; i < 16 * 128; i += 256) {
    int e = i >> 7, c = i & 127;
    int r = c >> 6, cc = c & 63;
    sA[e * 128 + c] = sH[e][(r ? 5 : 1) * 64 + cc];
    sA[2048 + e * 128 + c] = sH[e][(r ? 7 : 3) * 64 + cc];
  }
  __syncthreads();
  gemm16<4, 16, 2>(sA, 128, 2048, wpC2M1, nullptr, 0,
                   [&](int s, int e, int n, float v) {
                     sYm[s * 4096 + e * 256 + n] = (_Float16)v;
                   });
  __syncthreads();
  for (int i = t; i < 16 * 128; i += 256) {
    int e = i >> 7, c = i & 127;
    int r = c >> 6, cc = c & 63;
    float yr0 = (float)sYm[e * 256 + c],        yi0 = (float)sYm[e * 256 + 128 + c];
    float yr1 = (float)sYm[4096 + e * 256 + c], yi1 = (float)sYm[4096 + e * 256 + 128 + c];
    sV[e][(r ? 5 : 1) * 64 + cc] = (_Float16)(yr0 - yi1);
    sV[e][(r ? 7 : 3) * 64 + cc] = (_Float16)(yr1 + yi0);
  }
  __syncthreads();

  // m=2
  for (int i = t; i < 16 * 64; i += 256) {
    int e = i >> 6, c = i & 63;
    sA[e * 64 + c] = sH[e][4 * 64 + c];
    sA[1024 + e * 64 + c] = sH[e][8 * 64 + c];
  }
  __syncthreads();
  gemm16<2, 8, 2>(sA, 64, 1024, wpC2M2, nullptr, 0,
                  [&](int s, int e, int n, float v) {
                    sYm[s * 4096 + e * 256 + n] = (_Float16)v;
                  });
  __syncthreads();
  for (int i = t; i < 16 * 64; i += 256) {
    int e = i >> 6, cc = i & 63;
    float yr0 = (float)sYm[e * 256 + cc],        yi0 = (float)sYm[e * 256 + 64 + cc];
    float yr1 = (float)sYm[4096 + e * 256 + cc], yi1 = (float)sYm[4096 + e * 256 + 64 + cc];
    sV[e][4 * 64 + cc] = (_Float16)(yr0 - yi1);
    sV[e][8 * 64 + cc] = (_Float16)(yr1 + yi0);
  }
  __syncthreads();

  for (int i = t; i < 16 * 81; i += 256) sWig[i] = gWigG[(size_t)e0 * 81 + i];
  __syncthreads();

  // alpha-scale + wigner^T + atomic scatter into node accumulator
  for (int i = t; i < 16 * 64; i += 256) {
    int e = i >> 6, c = i & 63;
    float vv[9];
#pragma unroll
    for (int j = 0; j < 9; ++j) vv[j] = (float)sV[e][j * 64 + c];
    float a = sAl[e][c >> 4];
    float* np = gNode + (size_t)sDst[e] * 576 + c;
#pragma unroll
    for (int ii = 0; ii < 9; ++ii) {
      float s = 0.f;
#pragma unroll
      for (int j = 0; j < 9; ++j) s += sWig[e * 81 + j * 9 + ii] * vv[j];
      atomicAdd(np + ii * 64, a * s);
    }
  }
}

// x_new = x + so3_linear(node_accum, proj_w, proj_b)
__global__ __launch_bounds__(256) void node_proj_kernel(
    const float* __restrict__ gNode, const float* __restrict__ xin,
    const _Float16* __restrict__ wpProj, const float* __restrict__ projb,
    float* __restrict__ gXnew) {
  __shared__ _Float16 sA[16 * 576];
  const int t = threadIdx.x, n0 = blockIdx.x * 16;
  for (int i = t; i < 16 * 576; i += 256) sA[i] = (_Float16)gNode[(size_t)n0 * 576 + i];
  __syncthreads();
  gemm16<2, 4, 9>(sA, 576, 64, wpProj, kLof, 256, [&](int k, int m, int n, float v) {
    size_t idx = (size_t)(n0 + m) * 576 + k * 64 + n;
    gXnew[idx] = xin[idx] + v + (k == 0 ? projb[n] : 0.f);
  });
}

// out = x_new + so3_linear(sep_s2_act(gate(h2), so3_linear(h2, w1, b1)), w2, b2)
__global__ __launch_bounds__(256) void ffn_kernel(
    const float* __restrict__ gH2, const float* __restrict__ gXnew,
    const _Float16* __restrict__ wpGate, const float* __restrict__ gateb,
    const _Float16* __restrict__ wpF1, const float* __restrict__ f1b,
    const _Float16* __restrict__ wpF2, const float* __restrict__ f2b,
    const float* __restrict__ toG, const float* __restrict__ fromG,
    float* __restrict__ out) {
  __shared__ _Float16 sHin[16 * 576];
  __shared__ _Float16 sMid[16 * 576];
  __shared__ _Float16 sT[16 * 576];
  __shared__ _Float16 sGt[16 * 64];
  __shared__ float sTG[144], sFG[144];
  const int t = threadIdx.x, n0 = blockIdx.x * 16;
  for (int i = t; i < 16 * 576; i += 256) sHin[i] = (_Float16)gH2[(size_t)n0 * 576 + i];
  for (int i = t; i < 144; i += 256) { sTG[i] = toG[i]; sFG[i] = fromG[i]; }
  __syncthreads();
  gemm16<2, 4, 1>(sHin, 576, 0, wpGate, nullptr, 0, [&](int, int m, int n, float v) {
    sGt[m * 64 + n] = (_Float16)(v + gateb[n]);
  });
  gemm16<2, 4, 9>(sHin, 576, 64, wpF1, kLof, 256, [&](int k, int m, int n, float v) {
    sMid[m * 576 + k * 64 + n] = (_Float16)(v + (k == 0 ? f1b[n] : 0.f));
  });
  __syncthreads();
  for (int i = t; i < 16 * 64; i += 256) {
    int m = i >> 6, c = i & 63;
    float gv[16];
#pragma unroll
    for (int g = 0; g < 16; ++g) {
      float s = 0.f;
#pragma unroll
      for (int k = 0; k < 9; ++k) s += sTG[g * 9 + k] * (float)sMid[m * 576 + k * 64 + c];
      gv[g] = s * sigf(s);
    }
    float g0 = (float)sGt[m * 64 + c];
    sT[m * 576 + c] = (_Float16)(g0 * sigf(g0));
#pragma unroll
    for (int k = 1; k < 9; ++k) {
      float s = 0.f;
#pragma unroll
      for (int g = 0; g < 16; ++g) s += sFG[g * 9 + k] * gv[g];
      sT[m * 576 + k * 64 + c] = (_Float16)s;
    }
  }
  __syncthreads();
  gemm16<2, 4, 9>(sT, 576, 64, wpF2, kLof, 256, [&](int k, int m, int n, float v) {
    size_t idx = (size_t)(n0 + m) * 576 + k * 64 + n;
    out[idx] = gXnew[idx] + v + (k == 0 ? f2b[n] : 0.f);
  });
}

// ---------------------------------------------------------------------------
extern "C" void kernel_launch(void* const* d_in, const int* in_sizes, int n_in,
                              void* d_out, int out_size, void* d_ws, size_t ws_size,
                              hipStream_t stream) {
  (void)n_in; (void)out_size; (void)ws_size;
  // d_in order = setup_inputs() dict flattening: 33 params, x, edge_distance,
  // edge_index, wigner.
  const float* n1_w0 = (const float*)d_in[0];
  const float* n1_b0 = (const float*)d_in[1];
  const float* n1_wl = (const float*)d_in[2];
  const float* n2_w0 = (const float*)d_in[3];
  const float* n2_b0 = (const float*)d_in[4];
  const float* n2_wl = (const float*)d_in[5];
  const float* rad_b1 = (const float*)d_in[7];
  const float* rad_b2 = (const float*)d_in[9];
  const float* rad_b3 = (const float*)d_in[11];
  const float* c1_b0 = (const float*)d_in[13];
  const float* aln_w = (const float*)d_in[16];
  const float* aln_b = (const float*)d_in[17];
  const float* alpha_dot = (const float*)d_in[18];
  const float* c2_b0 = (const float*)d_in[20];
  const float* proj_b = (const float*)d_in[24];
  const float* gate_b = (const float*)d_in[26];
  const float* ffn_b1 = (const float*)d_in[28];
  const float* ffn_b2 = (const float*)d_in[30];
  const float* to_grid = (const float*)d_in[31];
  const float* from_grid = (const float*)d_in[32];
  const float* x = (const float*)d_in[33];
  const float* edist = (const float*)d_in[34];
  const int* eidx = (const int*)d_in[35];
  const float* wig = (const float*)d_in[36];

  const int nE = in_sizes[35] / 2;
  const int nN = in_sizes[33] / 576;
  const int* srcp = eidx;
  const int* dstp = eidx + nE;

  char* Bws = (char*)d_ws;
  _Float16* wpRad1 = (_Float16*)(Bws + 0);
  _Float16* wpRad2 = (_Float16*)(Bws + 8192);
  _Float16* wpRad3 = (_Float16*)(Bws + 16384);
  _Float16* wpC1W0 = (_Float16*)(Bws + 114688);
  _Float16* wpC1M1 = (_Float16*)(Bws + 409600);
  _Float16* wpC1M2 = (_Float16*)(Bws + 540672);
  _Float16* wpC2W0 = (_Float16*)(Bws + 573440);
  _Float16* wpC2M1 = (_Float16*)(Bws + 647168);
  _Float16* wpC2M2 = (_Float16*)(Bws + 712704);
  _Float16* wpGate = (_Float16*)(Bws + 729088);
  _Float16* wpProj = (_Float16*)(Bws + 737280);
  _Float16* wpF1   = (_Float16*)(Bws + 761856);
  _Float16* wpF2   = (_Float16*)(Bws + 786432);
  size_t off = (size_t)1 << 20;
  _Float16* gRad = (_Float16*)(Bws + off);  off += (size_t)nE * 768 * 2;
  float* gH1     = (float*)(Bws + off);     off += (size_t)nN * 576 * 4;
  _Float16* gHmsg = (_Float16*)(Bws + off); off += (size_t)nE * 576 * 2;
  float* gLogits = (float*)(Bws + off);     off += (size_t)nE * 4 * 4;
  float* gEx     = (float*)(Bws + off);     off += (size_t)nE * 4 * 4;
  float* gMx     = (float*)(Bws + off);     off += (size_t)nN * 4 * 4;
  float* gDen    = (float*)(Bws + off);     off += (size_t)nN * 4 * 4;
  float* gNode   = (float*)(Bws + off);     off += (size_t)nN * 576 * 4;
  float* gXnew   = (float*)(Bws + off);     off += (size_t)nN * 576 * 4;

  auto pack = [&](const void* W, _Float16* o, int K, int N) {
    int tot = K * N;
    pack_w_kernel<<<(tot + 255) / 256, 256, 0, stream>>>((const float*)W, o, K, N);
  };
  pack(d_in[6], wpRad1, 64, 64);
  pack(d_in[8], wpRad2, 64, 64);
  pack(d_in[10], wpRad3, 64, 768);
  pack(d_in[12], wpC1W0, 384, 384);
  pack(d_in[14], wpC1M1, 256, 256);
  pack(d_in[15], wpC1M2, 128, 128);
  pack(d_in[19], wpC2W0, 192, 192);
  pack(d_in[21], wpC2M1, 128, 256);
  pack(d_in[22], wpC2M2, 64, 128);
  pack(d_in[25], wpGate, 64, 64);
  for (int l = 0; l < 3; ++l) {
    pack((const float*)d_in[23] + l * 4096, wpProj + l * 4096, 64, 64);
    pack((const float*)d_in[27] + l * 4096, wpF1 + l * 4096, 64, 64);
    pack((const float*)d_in[29] + l * 4096, wpF2 + l * 4096, 64, 64);
  }

  init_kernel<<<(nN * 576 + 255) / 256, 256, 0, stream>>>(gNode, gMx, gDen, nN);
  eqnorm_kernel<<<nN, 64, 0, stream>>>(x, gH1, n1_w0, n1_b0, n1_wl);
  radial_kernel<<<nE / 16, 256, 0, stream>>>(edist, wpRad1, wpRad2, wpRad3, rad_b1,
                                             rad_b2, rad_b3, gRad);
  edge_pass_a<<<nE / 16, 256, 0, stream>>>(gH1, gRad, wig, srcp, dstp, wpC1W0, wpC1M1,
                                           wpC1M2, c1_b0, to_grid, from_grid, aln_w,
                                           aln_b, alpha_dot, gHmsg, gLogits, gMx);
  softmax_edge_kernel<<<(nE * 4 + 255) / 256, 256, 0, stream>>>(gLogits, gMx, dstp,
                                                                gEx, gDen, nE);
  edge_pass_b<<<nE / 16, 256, 0, stream>>>(gHmsg, wig, dstp, wpC2W0, wpC2M1, wpC2M2,
                                           c2_b0, gEx, gDen, gNode);
  node_proj_kernel<<<nN / 16, 256, 0, stream>>>(gNode, x, wpProj, proj_b, gXnew);
  eqnorm_kernel<<<nN, 64, 0, stream>>>(gXnew, gH1, n2_w0, n2_b0, n2_wl);
  ffn_kernel<<<nN / 16, 256, 0, stream>>>(gH1, gXnew, wpGate, gate_b, wpF1, ffn_b1,
                                          wpF2, ffn_b2, to_grid, from_grid,
                                          (float*)d_out);
}
